// Llama4Attention_33913061769721
// MI455X (gfx1250) — matrix operand order repaired and checked by reading, not compiled, not run on hardware
//
#include <hip/hip_runtime.h>

typedef unsigned short u16;
typedef unsigned int   u32;
typedef __attribute__((ext_vector_type(16))) __bf16 v16bf;
typedef __attribute__((ext_vector_type(8)))  float  v8f;

union Frag {
  v16bf v;
  uint4 q[2];
  u16   h[16];
};

__device__ inline u16 f2bf(float f) {
  u32 u = __float_as_uint(f);
  u32 r = u + 0x7FFFu + ((u >> 16) & 1u);   // round-to-nearest-even
  return (u16)(r >> 16);
}

// ---------------------------------------------------------------------------
// fp32 -> bf16 conversion, 8 elements / thread
// ---------------------------------------------------------------------------
__global__ __launch_bounds__(256) void cvt_bf16_kernel(
    const float* __restrict__ src, u16* __restrict__ dst, int n8)
{
  int i = blockIdx.x * 256 + threadIdx.x;
  if (i >= n8) return;
  const float4* s = (const float4*)src + (size_t)i * 2;
  float4 a = s[0], b = s[1];
  union { u16 h[8]; uint4 q; } o;
  o.h[0] = f2bf(a.x); o.h[1] = f2bf(a.y); o.h[2] = f2bf(a.z); o.h[3] = f2bf(a.w);
  o.h[4] = f2bf(b.x); o.h[5] = f2bf(b.y); o.h[6] = f2bf(b.z); o.h[7] = f2bf(b.w);
  ((uint4*)dst)[i] = o.q;
}

// ---------------------------------------------------------------------------
// C(MxN, f32) = A(MxK, bf16 row-major) * Bt(NxK, bf16 row-major)^T
// block = 128 threads (4 waves, 2x2), wave tile = 64x64 (16 WMMAs / k-step),
// block tile = 128x128. 8 fragment loads feed 16 WMMAs -> 2x the arithmetic
// intensity of a 32x32 wave tile.
// ---------------------------------------------------------------------------
__global__ __launch_bounds__(128) void gemm_bf16_wmma(
    const u16* __restrict__ A, const u16* __restrict__ Bt, float* __restrict__ C,
    int M, int N, int K)
{
  const int lane = threadIdx.x & 31;
  const int wave = threadIdx.x >> 5;
  const int m_base = blockIdx.y * 128 + (wave >> 1) * 64;
  const int n_base = blockIdx.x * 128 + (wave & 1) * 64;

  // A-frag (16x32): lanes 0-15 row M=l, K 0-15 ; lanes 16-31 K 16-31
  const int akb = (lane & 16) ? 16 : 0;
  // B-frag (32x16): lanes 0-15 col N=l, K 0-15 ; lanes 16-31 K 16-31
  const int bkb = (lane & 16) ? 16 : 0;

  const u16* ap[4];
  const u16* bp[4];
#pragma unroll
  for (int i = 0; i < 4; ++i) {
    ap[i] = A  + (size_t)(m_base + i * 16 + (lane & 15)) * K + akb;
    bp[i] = Bt + (size_t)(n_base + i * 16 + (lane & 15)) * K + bkb;
  }

  v8f acc[4][4];
#pragma unroll
  for (int i = 0; i < 4; ++i)
#pragma unroll
    for (int j = 0; j < 4; ++j)
#pragma unroll
      for (int e = 0; e < 8; ++e) acc[i][j][e] = 0.f;

  for (int k0 = 0; k0 < K; k0 += 32) {
    Frag af[4], bf[4];
#pragma unroll
    for (int i = 0; i < 4; ++i) {
      __builtin_prefetch(ap[i] + k0 + 128, 0, 1);   // global_prefetch_b8
      af[i].q[0] = *(const uint4*)(ap[i] + k0);
      af[i].q[1] = *(const uint4*)(ap[i] + k0 + 8);
    }
#pragma unroll
    for (int j = 0; j < 4; ++j) {
      __builtin_prefetch(bp[j] + k0 + 128, 0, 1);
      bf[j].q[0] = *(const uint4*)(bp[j] + k0);
      bf[j].q[1] = *(const uint4*)(bp[j] + k0 + 8);
    }
#pragma unroll
    for (int i = 0; i < 4; ++i)
#pragma unroll
      for (int j = 0; j < 4; ++j)
        acc[i][j] = __builtin_amdgcn_wmma_f32_16x16x32_bf16(
            false, af[i].v, false, bf[j].v, (short)0, acc[i][j], false, false);
  }

  // C/D layout: lanes 0-15 -> rows m..m+7 (VGPR e = row e), lanes 16-31 -> rows m+8..m+15
  const int rb = (lane & 16) ? 8 : 0;
  const int cc = lane & 15;
#pragma unroll
  for (int i = 0; i < 4; ++i)
#pragma unroll
    for (int j = 0; j < 4; ++j)
#pragma unroll
      for (int e = 0; e < 8; ++e)
        C[(size_t)(m_base + i * 16 + rb + e) * N + (n_base + j * 16 + cc)] = acc[i][j][e];
}

// ---------------------------------------------------------------------------
// RoPE (before norm, as in reference) + RMSNorm over flattened head dims,
// then emit bf16 Q (T,16,128), K (T,4,128), V (T,4,128). One block per token.
// ---------------------------------------------------------------------------
__device__ inline float block_sum256(float v, float* red) {
  __syncthreads();                              // protect red[] reuse
#pragma unroll
  for (int o = 16; o >= 1; o >>= 1) v += __shfl_xor(v, o, 32);
  const int tid = threadIdx.x;
  if ((tid & 31) == 0) red[tid >> 5] = v;
  __syncthreads();
  if (tid == 0) {
    float s = 0.f;
    for (int i = 0; i < 8; ++i) s += red[i];
    red[0] = s;
  }
  __syncthreads();
  return red[0];
}

__global__ __launch_bounds__(256) void rope_norm_kernel(
    const int* __restrict__ positions, const float* __restrict__ qkv,
    const float* __restrict__ qw, const float* __restrict__ kw,
    u16* __restrict__ qo, u16* __restrict__ ko, u16* __restrict__ vo)
{
  __shared__ float red[8];
  const int t   = blockIdx.x;
  const int tid = threadIdx.x;
  const float pos = (float)positions[t];
  const float* row = qkv + (size_t)t * 3072;

  // ---- Q: 16 heads x 64 rotation pairs = 1024 pairs; 4 per thread
  float y[8]; int hh[4], ii[4];
  float ssq = 0.f;
#pragma unroll
  for (int r = 0; r < 4; ++r) {
    int p = tid + r * 256;
    int h = p >> 6, i = p & 63;
    hh[r] = h; ii[r] = i;
    float freq = __expf(-9.210340372f * (float)(2 * i) * (1.0f / 128.0f)); // theta^(-2i/128)
    float sn, cs;
    sincosf(pos * freq, &sn, &cs);
    float x1 = row[h * 128 + i];
    float x2 = row[h * 128 + 64 + i];
    float y1 = x1 * cs - x2 * sn;
    float y2 = x2 * cs + x1 * sn;
    y[2 * r] = y1; y[2 * r + 1] = y2;
    ssq += y1 * y1 + y2 * y2;
  }
  float tot = block_sum256(ssq, red);
  float sc  = rsqrtf(tot * (1.0f / 2048.0f) + 1e-5f);
#pragma unroll
  for (int r = 0; r < 4; ++r) {
    int i1 = hh[r] * 128 + ii[r];
    int i2 = i1 + 64;
    qo[(size_t)t * 2048 + i1] = f2bf(y[2 * r]     * sc * qw[i1]);
    qo[(size_t)t * 2048 + i2] = f2bf(y[2 * r + 1] * sc * qw[i2]);
  }

  // ---- K: 4 heads x 64 pairs = 256 pairs; 1 per thread
  {
    const float* krow = row + 2048;
    int h = tid >> 6, i = tid & 63;
    float freq = __expf(-9.210340372f * (float)(2 * i) * (1.0f / 128.0f));
    float sn, cs;
    sincosf(pos * freq, &sn, &cs);
    float x1 = krow[h * 128 + i];
    float x2 = krow[h * 128 + 64 + i];
    float y1 = x1 * cs - x2 * sn;
    float y2 = x2 * cs + x1 * sn;
    float tk  = block_sum256(y1 * y1 + y2 * y2, red);
    float sck = rsqrtf(tk * (1.0f / 512.0f) + 1e-5f);
    int i1 = h * 128 + i, i2 = i1 + 64;
    ko[(size_t)t * 512 + i1] = f2bf(y1 * sck * kw[i1]);
    ko[(size_t)t * 512 + i2] = f2bf(y2 * sck * kw[i2]);
  }

  // ---- V passthrough to bf16
  {
    const float* vrow = row + 2560;
#pragma unroll
    for (int r = 0; r < 2; ++r) {
      int i = tid + r * 256;
      vo[(size_t)t * 512 + i] = f2bf(vrow[i]);
    }
  }
}

// ---------------------------------------------------------------------------
// Causal flash attention. grid = (T/64, H). block = 128 (4 waves x 16 q-rows).
// Q (T,16,128) bf16, K/V (T,4,128) bf16, O = attn (T, 2048) bf16.
// ---------------------------------------------------------------------------
#define ATT_SCALE 0.08838834764831845f
#define NEGV (-1.0e30f)

__global__ __launch_bounds__(128) void flash_attn_kernel(
    const u16* __restrict__ Q, const u16* __restrict__ Kmat,
    const u16* __restrict__ V, u16* __restrict__ O)
{
  __shared__ u16 Vt[128 * 32];      // V^T tile: [d][kv], 8KB
  __shared__ u16 Pt[4][16 * 32];    // per-wave P tile, 4KB

  const int h    = blockIdx.y;
  const int g    = h >> 2;          // kv head group (H/HKV = 4)
  const int q0   = blockIdx.x * 64;
  const int tid  = threadIdx.x;
  const int lane = tid & 31;
  const int wave = tid >> 5;
  const int qw   = q0 + wave * 16;

  // Q A-fragments: 16 rows x 128 = 4 k-slices of 32
  Frag aq[4];
  {
    const int row = qw + (lane & 15);
    const int kb  = (lane & 16) ? 16 : 0;
    const u16* qp = Q + ((size_t)row * 16 + h) * 128 + kb;
#pragma unroll
    for (int d = 0; d < 4; ++d) {
      aq[d].q[0] = *(const uint4*)(qp + d * 32);
      aq[d].q[1] = *(const uint4*)(qp + d * 32 + 8);
    }
  }

  v8f o_acc[8];
#pragma unroll
  for (int t = 0; t < 8; ++t)
#pragma unroll
    for (int j = 0; j < 8; ++j) o_acc[t][j] = 0.f;

  float mrow[8], lrow[8];
#pragma unroll
  for (int j = 0; j < 8; ++j) { mrow[j] = -3.0e38f; lrow[j] = 0.f; }

  const int rbase = qw + ((lane & 16) ? 8 : 0);
  const int colq  = lane & 15;

  // all waves iterate the same kv range -> barriers stay uniform
  for (int c0 = 0; c0 < q0 + 64; c0 += 32) {
    __syncthreads();
    // stage V^T: Vt[d*32+r] = V[c0+r][g][d]
    for (int idx = tid; idx < 32 * 128; idx += 128) {
      int r = idx >> 7, d = idx & 127;
      Vt[d * 32 + r] = V[((size_t)(c0 + r) * 4 + g) * 128 + d];
    }
    __syncthreads();

    // S(16x32) = Q * K^T, contraction over D=128 in 4 WMMA k-steps
    v8f s0, s1;
#pragma unroll
    for (int j = 0; j < 8; ++j) { s0[j] = 0.f; s1[j] = 0.f; }
    {
      const int bkb  = (lane & 16) ? 16 : 0;
      const u16* k0p = Kmat + ((size_t)(c0 + colq) * 4 + g) * 128 + bkb;
      const u16* k1p = k0p + (size_t)16 * 4 * 128;
#pragma unroll
      for (int d = 0; d < 4; ++d) {
        Frag b0, b1;
        b0.q[0] = *(const uint4*)(k0p + d * 32);
        b0.q[1] = *(const uint4*)(k0p + d * 32 + 8);
        b1.q[0] = *(const uint4*)(k1p + d * 32);
        b1.q[1] = *(const uint4*)(k1p + d * 32 + 8);
        s0 = __builtin_amdgcn_wmma_f32_16x16x32_bf16(false, aq[d].v, false, b0.v, (short)0, s0, false, false);
        s1 = __builtin_amdgcn_wmma_f32_16x16x32_bf16(false, aq[d].v, false, b1.v, (short)0, s1, false, false);
      }
    }

    // online softmax update (row = rbase+j, 16 cols of a row live across 16 lanes)
#pragma unroll
    for (int j = 0; j < 8; ++j) {
      const int rowj = rbase + j;
      float v0 = s0[j] * ATT_SCALE;
      float v1 = s1[j] * ATT_SCALE;
      if (c0 + colq > rowj)      v0 = NEGV;
      if (c0 + 16 + colq > rowj) v1 = NEGV;
      float mx = fmaxf(v0, v1);
#pragma unroll
      for (int o = 1; o < 16; o <<= 1) mx = fmaxf(mx, __shfl_xor(mx, o, 32));
      const float mnew = fmaxf(mrow[j], mx);
      const float p0 = __expf(v0 - mnew);
      const float p1 = __expf(v1 - mnew);
      float ps = p0 + p1;
#pragma unroll
      for (int o = 1; o < 16; o <<= 1) ps += __shfl_xor(ps, o, 32);
      const float f = __expf(mrow[j] - mnew);
      lrow[j] = lrow[j] * f + ps;
      mrow[j] = mnew;
#pragma unroll
      for (int t = 0; t < 8; ++t) o_acc[t][j] *= f;
      s0[j] = p0; s1[j] = p1;
    }

    // P -> LDS bf16 (C-layout scatter), then reload as A-fragment
    {
      u16* pw = Pt[wave];
      const int rb = (lane & 16) ? 8 : 0;
#pragma unroll
      for (int j = 0; j < 8; ++j) {
        pw[(rb + j) * 32 + colq]      = f2bf(s0[j]);
        pw[(rb + j) * 32 + colq + 16] = f2bf(s1[j]);
      }
    }
    asm volatile("s_wait_dscnt 0" ::: "memory");   // DS RAW fence (same wave)

    Frag pa;
    {
      const u16* pw = Pt[wave];
      const int prow = lane & 15;
      const int kb   = (lane & 16) ? 16 : 0;
      pa.q[0] = *(const uint4*)(pw + prow * 32 + kb);
      pa.q[1] = *(const uint4*)(pw + prow * 32 + kb + 8);
    }

    // O(16x128) += P(16x32) * V(32x128): 8 n-tiles, B-frags from Vt in LDS
    {
      const int bkb = (lane & 16) ? 16 : 0;
#pragma unroll
      for (int t = 0; t < 8; ++t) {
        const u16* vp = Vt + (t * 16 + colq) * 32 + bkb;
        Frag b;
        b.q[0] = *(const uint4*)(vp);
        b.q[1] = *(const uint4*)(vp + 8);
        o_acc[t] = __builtin_amdgcn_wmma_f32_16x16x32_bf16(
            false, pa.v, false, b.v, (short)0, o_acc[t], false, false);
      }
    }
  }

  // epilogue: divide by softmax denominator, store attn bf16 (T, H*D)
  {
    const int rb = (lane & 16) ? 8 : 0;
#pragma unroll
    for (int j = 0; j < 8; ++j) {
      const int row = qw + rb + j;
      const float inv = 1.0f / lrow[j];
#pragma unroll
      for (int t = 0; t < 8; ++t)
        O[(size_t)row * 2048 + h * 128 + t * 16 + colq] = f2bf(o_acc[t][j] * inv);
    }
  }
}

// ---------------------------------------------------------------------------
// launch
// ---------------------------------------------------------------------------
extern "C" void kernel_launch(void* const* d_in, const int* in_sizes, int n_in,
                              void* d_out, int out_size, void* d_ws, size_t ws_size,
                              hipStream_t stream) {
  (void)in_sizes; (void)n_in; (void)out_size; (void)ws_size;
  const int*   positions = (const int*)  d_in[0];
  const float* hidden    = (const float*)d_in[1];
  const float* w_qkv     = (const float*)d_in[2];
  const float* w_o       = (const float*)d_in[3];
  const float* q_norm_w  = (const float*)d_in[4];
  const float* k_norm_w  = (const float*)d_in[5];
  float* out = (float*)d_out;
  char*  ws  = (char*)d_ws;

  // workspace layout (regions reused once their producer/consumer is done):
  u16*   hid_bf  = (u16*)(ws + 0);             // 16 MB (dead after qkv GEMM)
  u16*   wqkv_bf = (u16*)(ws + 16777216);      // 12 MB (dead after qkv GEMM)
  u16*   wo_bf   = (u16*)(ws + 29360128);      //  8 MB (live to the end)
  float* qkv     = (float*)(ws + 37748736);    // 48 MB (dead after rope)
  u16*   qbf     = (u16*)(ws + 0);             // 16 MB (overlays hid_bf)
  u16*   kbf     = (u16*)(ws + 16777216);      //  4 MB (overlays wqkv_bf)
  u16*   vbf     = (u16*)(ws + 20971520);      //  4 MB
  u16*   attnbf  = (u16*)(ws + 37748736);      // 16 MB (overlays qkv)

  cvt_bf16_kernel<<<4096, 256, 0, stream>>>(hidden, hid_bf, 1048576); // 4096*2048/8
  cvt_bf16_kernel<<<3072, 256, 0, stream>>>(w_qkv, wqkv_bf, 786432);  // 3072*2048/8
  cvt_bf16_kernel<<<2048, 256, 0, stream>>>(w_o,   wo_bf,   524288);  // 2048*2048/8

  // qkv = hidden @ w_qkv^T : (4096 x 2048) x (3072 x 2048)^T, block tile 128x128
  gemm_bf16_wmma<<<dim3(24, 32), 128, 0, stream>>>(hid_bf, wqkv_bf, qkv, 4096, 3072, 2048);

  rope_norm_kernel<<<4096, 256, 0, stream>>>(positions, qkv, q_norm_w, k_norm_w, qbf, kbf, vbf);

  flash_attn_kernel<<<dim3(64, 16), 128, 0, stream>>>(qbf, kbf, vbf, attnbf);

  // out = attn @ w_o^T : (4096 x 2048) x (2048 x 2048)^T, block tile 128x128
  gemm_bf16_wmma<<<dim3(16, 32), 128, 0, stream>>>(attnbf, wo_bf, out, 4096, 2048, 2048);
}
